// FactorizedSSMBlock_31610959299143
// MI455X (gfx1250) — matrix operand order repaired and checked by reading, not compiled, not run on hardware
//
#include <hip/hip_runtime.h>
#include <cstdint>
#include <cstddef>

// ---------------- problem constants ----------------
#define L_SEQ   4096
#define NROWS   32768          // 8 sequences * 4096
#define DMODEL  256
#define DINNER  512
#define NH      8
#define HD      64
#define NSTATE  64
#define CONVD   640            // DINNER + 2*NSTATE
#define NPROJ   1152           // z(512) + xBC(640); dt(8) handled separately
#define DINPROJ 1160
#define QCH     64             // scan chunk length
#define NCHUNK  64             // 4096 / 64

typedef __attribute__((ext_vector_type(16))) _Float16 v16h;
typedef __attribute__((ext_vector_type(8)))  float    v8f;

// ---------------- WMMA helpers (gfx1250, wave32) ----------------
__device__ __forceinline__ v8f wmma_f16(v16h a, v16h b, v8f c) {
  return __builtin_amdgcn_wmma_f32_16x16x32_f16(false, a, false, b, (short)0, c,
                                                false, false);
}

// A fragment: 16x32 f16 tile, row-major source with leading dim lda.
// lane L: row = row0 + (L&15); halves i -> K = k0 + (L>>4)*16 + i (contiguous)
__device__ __forceinline__ v16h load_a_frag(const _Float16* A, int lda,
                                            int row0, int k0) {
  int lane = threadIdx.x & 31;
  int m  = lane & 15;
  int hi = lane >> 4;
  const _Float16* p0 = A + (size_t)(row0 + m) * lda + k0 + hi * 16;
  v16h a;
  ((uint4*)&a)[0] = *(const uint4*)p0;
  ((uint4*)&a)[1] = *(const uint4*)(p0 + 8);
  return a;
}

// B fragment for D = A * W^T : B_op[k, n] = W[col0+n][k], W row-major ldw.
// lane L: col = col0 + (L&15); halves i -> K = k0 + (L>>4)*16 + i (contiguous)
__device__ __forceinline__ v16h load_b_frag(const _Float16* W, int ldw,
                                            int col0, int k0) {
  int lane = threadIdx.x & 31;
  int n  = lane & 15;
  int hi = lane >> 4;
  const _Float16* p = W + (size_t)(col0 + n) * ldw + k0 + hi * 16;
  v16h b;
  ((uint4*)&b)[0] = ((const uint4*)p)[0];
  ((uint4*)&b)[1] = ((const uint4*)p)[1];
  return b;
}

__device__ __forceinline__ float silu_f(float x) {
  return x / (1.0f + __expf(-x));
}

// ---------------- small utility kernels ----------------
__global__ void f32_to_f16_kernel(const float* __restrict__ src,
                                  _Float16* __restrict__ dst, int n) {
  int i = blockIdx.x * 256 + threadIdx.x;
  if (i < n) dst[i] = (_Float16)src[i];
}

// LayerNorm over 256 + cast to f16. One row per block (256 threads).
__global__ void __launch_bounds__(256)
ln_cast_kernel(const float* __restrict__ x0, const float* __restrict__ x1,
               const float* __restrict__ w, const float* __restrict__ b,
               _Float16* __restrict__ Xn) {
  size_t row = blockIdx.x;
  const float* x = (row < (NROWS / 2)) ? x0 + row * DMODEL
                                       : x1 + (row - NROWS / 2) * DMODEL;
  int tid = threadIdx.x;
  __shared__ float red[256];
  float v = x[tid];
  red[tid] = v; __syncthreads();
  for (int s = 128; s > 0; s >>= 1) { if (tid < s) red[tid] += red[tid + s]; __syncthreads(); }
  float mu = red[0] * (1.0f / DMODEL);
  __syncthreads();
  float xc = v - mu;
  red[tid] = xc * xc; __syncthreads();
  for (int s = 128; s > 0; s >>= 1) { if (tid < s) red[tid] += red[tid + s]; __syncthreads(); }
  float inv = rsqrtf(red[0] * (1.0f / DMODEL) + 1e-5f);
  Xn[row * DMODEL + tid] = (_Float16)(xc * inv * w[tid] + b[tid]);
}

// ---------------- GEMM1: zxbcdt[:, :1152] = Xn @ W_in^T ----------------
// Strip-mined: each wave computes a 16x64 strip (A fragment reused 4x).
__global__ void __launch_bounds__(256)
gemm_in_kernel(const _Float16* __restrict__ Xn, const _Float16* __restrict__ Wh,
               float* __restrict__ Out) {
  int strip = blockIdx.x * 8 + (threadIdx.x >> 5);
  const int NS = NPROJ / 64;                 // 18 strips of 64 columns
  int mt = strip / NS, ns = strip % NS;
  int row0 = mt * 16, col0 = ns * 64;
  v8f acc[4] = {};
  for (int k0 = 0; k0 < DMODEL; k0 += 32) {
    v16h a = load_a_frag(Xn, DMODEL, row0, k0);
#pragma unroll
    for (int j = 0; j < 4; ++j) {
      v16h b = load_b_frag(Wh, DMODEL, col0 + j * 16, k0);
      acc[j] = wmma_f16(a, b, acc[j]);
    }
  }
  int lane = threadIdx.x & 31;
  int n = lane & 15, hi = lane >> 4;
#pragma unroll
  for (int j = 0; j < 4; ++j) {
    int col = col0 + j * 16 + n;
#pragma unroll
    for (int r = 0; r < 8; ++r) {
      size_t row = (size_t)row0 + hi * 8 + r;
      Out[row * NPROJ + col] = acc[j][r];
    }
  }
}

// ---------------- dt projection (8 cols) + softplus + log-decay ----------------
__global__ void __launch_bounds__(256)
dt_kernel(const _Float16* __restrict__ Xn, const _Float16* __restrict__ Wh,
          const float* __restrict__ dt_bias, const float* __restrict__ A_log,
          float* __restrict__ dtv, float* __restrict__ lav) {
  int idx = blockIdx.x * 256 + threadIdx.x;   // NROWS * NH
  if (idx >= NROWS * NH) return;
  int h = idx & 7;
  size_t row = (size_t)idx >> 3;
  const uint4* a4 = (const uint4*)(Xn + row * DMODEL);
  const uint4* w4 = (const uint4*)(Wh + (size_t)(NPROJ + h) * DMODEL);
  float acc = 0.0f;
  for (int k = 0; k < DMODEL / 8; ++k) {
    uint4 av = a4[k], wv = w4[k];
    const _Float16* ae = (const _Float16*)&av;
    const _Float16* we = (const _Float16*)&wv;
#pragma unroll
    for (int i = 0; i < 8; ++i) acc += (float)ae[i] * (float)we[i];
  }
  float x = acc + dt_bias[h];
  float dt = (x > 20.0f) ? x : log1pf(__expf(x));   // softplus
  dtv[idx] = dt;
  lav[idx] = -__expf(A_log[h]) * dt;                // log(dA)
}

// ---------------- depthwise causal conv(4) + SiLU, split into xh/B/C ----------------
__global__ void __launch_bounds__(256)
conv_kernel(const float* __restrict__ Z, const float* __restrict__ conv_w,
            const float* __restrict__ conv_b, _Float16* __restrict__ Xh,
            _Float16* __restrict__ Bh, _Float16* __restrict__ Ch) {
  size_t idx = (size_t)blockIdx.x * 256 + threadIdx.x;   // NROWS * CONVD
  if (idx >= (size_t)NROWS * CONVD) return;
  int c = (int)(idx % CONVD);
  size_t row = idx / CONVD;
  int l = (int)(row & (L_SEQ - 1));
  float acc = conv_b[c];
#pragma unroll
  for (int k = 0; k < 4; ++k) {
    int dl = l - 3 + k;
    if (dl >= 0) acc += Z[(row - 3 + k) * NPROJ + DINNER + c] * conv_w[c * 4 + k];
  }
  acc = silu_f(acc);
  _Float16 hv = (_Float16)acc;
  if (c < DINNER)            Xh[row * DINNER + c] = hv;
  else if (c < DINNER + 64)  Bh[row * NSTATE + (c - DINNER)] = hv;
  else                       Ch[row * NSTATE + (c - DINNER - 64)] = hv;
}

// ---------------- chunked SSM scan: one block per (batch, head) ----------------
// All four per-chunk matmuls (G = C B^T, Yintra = M.G @ X, Yinter = C H0^T,
// H_new = efull*H0 + (w.X)^T B) use WMMA with 128-bit LDS fragment loads.
__global__ void __launch_bounds__(256)
scan_kernel(const _Float16* __restrict__ Xh, const _Float16* __restrict__ Bh,
            const _Float16* __restrict__ Ch, const float* __restrict__ dtv,
            const float* __restrict__ lav, const float* __restrict__ Dp,
            float* __restrict__ Y) {
  const int bh = blockIdx.x;                 // 0..63
  const int b = bh >> 3, h = bh & 7;
  const int tid = threadIdx.x;
  const int wave = tid >> 5, lane = tid & 31;

  __shared__ __align__(16) _Float16 sC [QCH * NSTATE];  // [t][n]
  __shared__ __align__(16) _Float16 sB [QCH * NSTATE];  // [s][n]
  __shared__ __align__(16) _Float16 sBT[NSTATE * QCH];  // [n][s] (B transposed)
  __shared__ __align__(16) _Float16 sXT[HD * QCH];      // [p][t] (X transposed)
  __shared__ __align__(16) _Float16 sXW[HD * QCH];      // [p][s] = w[s]*X[s][p]
  __shared__ __align__(16) _Float16 sG [QCH * QCH];     // masked decay*G [t][s]
  __shared__ __align__(16) _Float16 sHh[HD * NSTATE];   // f16 mirror of state
  __shared__ float sH[2][HD * NSTATE];                  // state [p][n], ping-pong
  __shared__ float s_cs[QCH], s_dt[QCH], s_w[QCH], s_e[QCH];
  __shared__ float s_efull;

  for (int i = tid; i < HD * NSTATE; i += 256) { sH[0][i] = 0.0f; sHh[i] = (_Float16)0.0f; }
  const float Dh = Dp[h];
  const size_t rowbase = (size_t)b * L_SEQ;
  int cur = 0;
  __syncthreads();

  for (int c = 0; c < NCHUNK; ++c) {
    // ---- stage chunk into LDS (row-major + transposed copies) ----
    {
      int t = tid >> 2;                 // 0..63
      int seg = (tid & 3) * 16;         // 16 halves per thread
      size_t row = rowbase + (size_t)c * QCH + t;
      const _Float16* xp = Xh + row * DINNER + h * HD + seg;
      const _Float16* bp = Bh + row * NSTATE + seg;
#pragma unroll
      for (int i = 0; i < 16; ++i) sXT[(seg + i) * QCH + t] = xp[i];
#pragma unroll
      for (int i = 0; i < 16; ++i) sBT[(seg + i) * QCH + t] = bp[i];
      *(uint4*)&sB[t * NSTATE + seg]     = *(const uint4*)(bp);
      *(uint4*)&sB[t * NSTATE + seg + 8] = *(const uint4*)(bp + 8);
      *(uint4*)&sC[t * NSTATE + seg]     = *(const uint4*)(Ch + row * NSTATE + seg);
      *(uint4*)&sC[t * NSTATE + seg + 8] = *(const uint4*)(Ch + row * NSTATE + seg + 8);
    }
    if (tid < QCH) {
      size_t row = rowbase + (size_t)c * QCH + tid;
      s_dt[tid] = dtv[row * NH + h];
      s_cs[tid] = lav[row * NH + h];    // temp: raw log-decay
    }
    __syncthreads();
    if (tid == 0) {                     // serial cumsum (64 adds, cheap)
      float run = 0.0f;
      for (int t = 0; t < QCH; ++t) { run += s_cs[t]; s_cs[t] = run; }
      s_efull = __expf(run);
    }
    __syncthreads();
    if (tid < QCH) {
      float cs_end = s_cs[QCH - 1];
      s_e[tid] = __expf(s_cs[tid]);
      s_w[tid] = __expf(cs_end - s_cs[tid]) * s_dt[tid];
    }
    __syncthreads();
    // pre-scale X by w[s] into A-fragment layout: sXW[p][s] = w[s]*X[s][p]
    {
      int p = tid >> 2;
      int seg = (tid & 3) * 16;
#pragma unroll
      for (int i = 0; i < 16; ++i) {
        int s = seg + i;
        sXW[p * QCH + s] = (_Float16)((float)sXT[p * QCH + s] * s_w[s]);
      }
    }
    __syncthreads();

    // ---- G = C @ B^T, then mask/decay into sG (f16) ----
    for (int tile = wave; tile < 16; tile += 8) {
      int t0 = (tile >> 2) * 16, s0 = (tile & 3) * 16;
      v8f acc = {};
      for (int k0 = 0; k0 < NSTATE; k0 += 32) {
        v16h a  = load_a_frag(sC, NSTATE, t0, k0);
        v16h bf = load_b_frag(sB, NSTATE, s0, k0);   // B_op[k,s] = B[s][k]
        acc = wmma_f16(a, bf, acc);
      }
      int sj = lane & 15, hi = lane >> 4;
      int scol = s0 + sj;
#pragma unroll
      for (int r = 0; r < 8; ++r) {
        int trow = t0 + hi * 8 + r;
        float m = (scol <= trow)
                      ? __expf(s_cs[trow] - s_cs[scol]) * s_dt[scol] : 0.0f;
        sG[trow * QCH + scol] = (_Float16)(acc[r] * m);
      }
    }
    __syncthreads();

    // ---- Y = sG @ X  +  e_t * (C @ H0^T)  +  D*x ----
    for (int tile = wave; tile < 16; tile += 8) {
      int t0 = (tile >> 2) * 16, p0 = (tile & 3) * 16;
      v8f acc = {};
      v8f acc2 = {};
      for (int k0 = 0; k0 < QCH; k0 += 32) {
        v16h a  = load_a_frag(sG, QCH, t0, k0);
        v16h bf = load_b_frag(sXT, QCH, p0, k0);     // B_op[s,p] = X[s][p]
        acc = wmma_f16(a, bf, acc);
      }
      for (int k0 = 0; k0 < NSTATE; k0 += 32) {
        v16h a  = load_a_frag(sC, NSTATE, t0, k0);
        v16h bf = load_b_frag(sHh, NSTATE, p0, k0);  // B_op[n,p] = H0[p][n]
        acc2 = wmma_f16(a, bf, acc2);
      }
      int pj = lane & 15, hi = lane >> 4;
      int pcol = p0 + pj;
#pragma unroll
      for (int r = 0; r < 8; ++r) {
        int t = t0 + hi * 8 + r;
        float xv = (float)sXT[pcol * QCH + t];
        float y = acc[r] + s_e[t] * acc2[r] + Dh * xv;
        size_t row = rowbase + (size_t)c * QCH + t;
        Y[row * DINNER + h * HD + pcol] = y;
      }
    }
    __syncthreads();   // Y-phase readers of sHh must finish before H update

    // ---- H_new = e_full * H0 + (w . X)^T @ B ----
    int nxt = cur ^ 1;
    for (int tile = wave; tile < 16; tile += 8) {
      int p0 = (tile >> 2) * 16, n0 = (tile & 3) * 16;
      int nj = lane & 15, hi = lane >> 4;
      int ncol = n0 + nj;
      v8f acc;
#pragma unroll
      for (int r = 0; r < 8; ++r)
        acc[r] = s_efull * sH[cur][(p0 + hi * 8 + r) * NSTATE + ncol];
      for (int k0 = 0; k0 < QCH; k0 += 32) {
        v16h a  = load_a_frag(sXW, QCH, p0, k0);     // A_op[p,s] = w[s]*X[s][p]
        v16h bf = load_b_frag(sBT, QCH, n0, k0);     // B_op[s,n] = B[s][n]
        acc = wmma_f16(a, bf, acc);
      }
#pragma unroll
      for (int r = 0; r < 8; ++r) {
        int p = p0 + hi * 8 + r;
        sH[nxt][p * NSTATE + ncol] = acc[r];
        sHh[p * NSTATE + ncol] = (_Float16)acc[r];
      }
    }
    cur ^= 1;
    __syncthreads();
  }
}

// ---------------- gate * silu(z), RMSNorm(512), cast f16 ----------------
__global__ void __launch_bounds__(256)
gate_rms_kernel(const float* __restrict__ Yv, const float* __restrict__ Z,
                const float* __restrict__ rms_w, _Float16* __restrict__ Yg) {
  size_t row = blockIdx.x;
  int tid = threadIdx.x;
  __shared__ float red[256];
  float v[2]; float ss = 0.0f;
#pragma unroll
  for (int i = 0; i < 2; ++i) {
    int j = tid + i * 256;
    float z = Z[row * NPROJ + j];                 // z = zxbcdt[:, :512]
    float val = Yv[row * DINNER + j] * silu_f(z);
    v[i] = val; ss += val * val;
  }
  red[tid] = ss; __syncthreads();
  for (int s = 128; s > 0; s >>= 1) { if (tid < s) red[tid] += red[tid + s]; __syncthreads(); }
  float scale = rsqrtf(red[0] * (1.0f / DINNER) + 1e-5f);
#pragma unroll
  for (int i = 0; i < 2; ++i) {
    int j = tid + i * 256;
    Yg[row * DINNER + j] = (_Float16)(v[i] * scale * rms_w[j]);
  }
}

// ---------------- GEMM2: sp = Yg @ W_out^T + x_cat (16x64 strips) ----------------
__global__ void __launch_bounds__(256)
gemm_out_kernel(const _Float16* __restrict__ Yg, const _Float16* __restrict__ Wo,
                const float* __restrict__ x0, const float* __restrict__ x1,
                float* __restrict__ Sp, _Float16* __restrict__ Sph) {
  int strip = blockIdx.x * 8 + (threadIdx.x >> 5);
  int mt = strip / 4, ns = strip % 4;        // 256 cols = 4 strips of 64
  int row0 = mt * 16, col0 = ns * 64;
  v8f acc[4] = {};
  for (int k0 = 0; k0 < DINNER; k0 += 32) {
    v16h a = load_a_frag(Yg, DINNER, row0, k0);
#pragma unroll
    for (int j = 0; j < 4; ++j) {
      v16h b = load_b_frag(Wo, DINNER, col0 + j * 16, k0);
      acc[j] = wmma_f16(a, b, acc[j]);
    }
  }
  int lane = threadIdx.x & 31;
  int n = lane & 15, hi = lane >> 4;
#pragma unroll
  for (int j = 0; j < 4; ++j) {
    int col = col0 + j * 16 + n;
#pragma unroll
    for (int r = 0; r < 8; ++r) {
      size_t row = (size_t)row0 + hi * 8 + r;
      const float* xr = (row < (NROWS / 2)) ? x0 + row * DMODEL
                                            : x1 + (row - NROWS / 2) * DMODEL;
      float v = acc[j][r] + xr[col];
      Sp[row * DMODEL + col] = v;
      Sph[row * DMODEL + col] = (_Float16)v;
    }
  }
}

// ------ GEMM3: h = silu(concat(f_self, f_partner) @ W1^T + b1) (16x64 strips) ------
__global__ void __launch_bounds__(256)
gemm_mlp1_kernel(const _Float16* __restrict__ Sph, const _Float16* __restrict__ W1h,
                 const float* __restrict__ b1, _Float16* __restrict__ Hh) {
  int strip = blockIdx.x * 8 + (threadIdx.x >> 5);
  int mt = strip / 4, ns = strip % 4;
  int row0 = mt * 16, col0 = ns * 64;
  int row0p = row0 ^ (NROWS / 2);            // partner rows (swap halves)
  v8f acc[4] = {};
  for (int k0 = 0; k0 < 2 * DMODEL; k0 += 32) {
    v16h a = (k0 < DMODEL) ? load_a_frag(Sph, DMODEL, row0, k0)
                           : load_a_frag(Sph, DMODEL, row0p, k0 - DMODEL);
#pragma unroll
    for (int j = 0; j < 4; ++j) {
      v16h b = load_b_frag(W1h, 2 * DMODEL, col0 + j * 16, k0);
      acc[j] = wmma_f16(a, b, acc[j]);
    }
  }
  int lane = threadIdx.x & 31;
  int n = lane & 15, hi = lane >> 4;
#pragma unroll
  for (int j = 0; j < 4; ++j) {
    int col = col0 + j * 16 + n;
#pragma unroll
    for (int r = 0; r < 8; ++r) {
      size_t row = (size_t)row0 + hi * 8 + r;
      Hh[row * DMODEL + col] = (_Float16)silu_f(acc[j][r] + b1[col]);
    }
  }
}

// ---------------- GEMM4: u = h @ W2^T + b2 + sp (16x64 strips) ----------------
__global__ void __launch_bounds__(256)
gemm_mlp2_kernel(const _Float16* __restrict__ Hh, const _Float16* __restrict__ W2h,
                 const float* __restrict__ b2, const float* __restrict__ Sp,
                 float* __restrict__ U) {
  int strip = blockIdx.x * 8 + (threadIdx.x >> 5);
  int mt = strip / 4, ns = strip % 4;
  int row0 = mt * 16, col0 = ns * 64;
  v8f acc[4] = {};
  for (int k0 = 0; k0 < DMODEL; k0 += 32) {
    v16h a = load_a_frag(Hh, DMODEL, row0, k0);
#pragma unroll
    for (int j = 0; j < 4; ++j) {
      v16h b = load_b_frag(W2h, DMODEL, col0 + j * 16, k0);
      acc[j] = wmma_f16(a, b, acc[j]);
    }
  }
  int lane = threadIdx.x & 31;
  int n = lane & 15, hi = lane >> 4;
#pragma unroll
  for (int j = 0; j < 4; ++j) {
    int col = col0 + j * 16 + n;
#pragma unroll
    for (int r = 0; r < 8; ++r) {
      size_t row = (size_t)row0 + hi * 8 + r;
      U[row * DMODEL + col] = acc[j][r] + b2[col] + Sp[row * DMODEL + col];
    }
  }
}

// ---------------- final LayerNorm(256) -> d_out ----------------
__global__ void __launch_bounds__(256)
final_ln_kernel(const float* __restrict__ U, const float* __restrict__ w,
                const float* __restrict__ b, float* __restrict__ Out) {
  size_t row = blockIdx.x;
  int tid = threadIdx.x;
  __shared__ float red[256];
  float v = U[row * DMODEL + tid];
  red[tid] = v; __syncthreads();
  for (int s = 128; s > 0; s >>= 1) { if (tid < s) red[tid] += red[tid + s]; __syncthreads(); }
  float mu = red[0] * (1.0f / DMODEL);
  __syncthreads();
  float xc = v - mu;
  red[tid] = xc * xc; __syncthreads();
  for (int s = 128; s > 0; s >>= 1) { if (tid < s) red[tid] += red[tid + s]; __syncthreads(); }
  float inv = rsqrtf(red[0] * (1.0f / DMODEL) + 1e-5f);
  Out[row * DMODEL + tid] = xc * inv * w[tid] + b[tid];
}

// ---------------- host-side orchestration ----------------
extern "C" void kernel_launch(void* const* d_in, const int* in_sizes, int n_in,
                              void* d_out, int out_size, void* d_ws, size_t ws_size,
                              hipStream_t stream) {
  (void)in_sizes; (void)n_in; (void)out_size; (void)ws_size;
  const float* x0      = (const float*)d_in[0];
  const float* x1      = (const float*)d_in[1];
  const float* ns_w    = (const float*)d_in[2];
  const float* ns_b    = (const float*)d_in[3];
  const float* W_in    = (const float*)d_in[4];
  const float* conv_w  = (const float*)d_in[5];
  const float* conv_b  = (const float*)d_in[6];
  const float* dt_bias = (const float*)d_in[7];
  const float* A_log   = (const float*)d_in[8];
  const float* D_param = (const float*)d_in[9];
  const float* rms_w   = (const float*)d_in[10];
  const float* W_out   = (const float*)d_in[11];
  const float* mlp_w1  = (const float*)d_in[12];
  const float* mlp_b1  = (const float*)d_in[13];
  const float* mlp_w2  = (const float*)d_in[14];
  const float* mlp_b2  = (const float*)d_in[15];
  const float* nt_w    = (const float*)d_in[16];
  const float* nt_b    = (const float*)d_in[17];
  float* out = (float*)d_out;

  // workspace carve-up (256B aligned)
  size_t off = 0;
  char* base = (char*)d_ws;
  auto carve = [&](size_t bytes) -> void* {
    off = (off + 255) & ~(size_t)255;
    void* p = base + off;
    off += bytes;
    return p;
  };
  _Float16* w_in_h = (_Float16*)carve((size_t)DINPROJ * DMODEL * 2);
  _Float16* w_out_h= (_Float16*)carve((size_t)DMODEL * DINNER * 2);
  _Float16* w1_h   = (_Float16*)carve((size_t)DMODEL * 2 * DMODEL * 2);
  _Float16* w2_h   = (_Float16*)carve((size_t)DMODEL * DMODEL * 2);
  _Float16* xn_h   = (_Float16*)carve((size_t)NROWS * DMODEL * 2);
  float*    zx     = (float*)   carve((size_t)NROWS * NPROJ * 4);
  float*    dtv    = (float*)   carve((size_t)NROWS * NH * 4);
  float*    lav    = (float*)   carve((size_t)NROWS * NH * 4);
  _Float16* xh_h   = (_Float16*)carve((size_t)NROWS * DINNER * 2);
  _Float16* bm_h   = (_Float16*)carve((size_t)NROWS * NSTATE * 2);
  _Float16* cm_h   = (_Float16*)carve((size_t)NROWS * NSTATE * 2);
  float*    ys     = (float*)   carve((size_t)NROWS * DINNER * 4);
  _Float16* yg_h   = (_Float16*)carve((size_t)NROWS * DINNER * 2);
  float*    sp_f   = (float*)   carve((size_t)NROWS * DMODEL * 4);
  _Float16* sp_h   = (_Float16*)carve((size_t)NROWS * DMODEL * 2);
  _Float16* hm_h   = (_Float16*)carve((size_t)NROWS * DMODEL * 2);
  float*    u_f    = (float*)   carve((size_t)NROWS * DMODEL * 4);

  // 1. weight conversions to f16
  f32_to_f16_kernel<<<(DINPROJ * DMODEL + 255) / 256, 256, 0, stream>>>(W_in, w_in_h, DINPROJ * DMODEL);
  f32_to_f16_kernel<<<(DMODEL * DINNER + 255) / 256, 256, 0, stream>>>(W_out, w_out_h, DMODEL * DINNER);
  f32_to_f16_kernel<<<(DMODEL * 2 * DMODEL + 255) / 256, 256, 0, stream>>>(mlp_w1, w1_h, DMODEL * 2 * DMODEL);
  f32_to_f16_kernel<<<(DMODEL * DMODEL + 255) / 256, 256, 0, stream>>>(mlp_w2, w2_h, DMODEL * DMODEL);

  // 2. input layernorm + cast
  ln_cast_kernel<<<NROWS, 256, 0, stream>>>(x0, x1, ns_w, ns_b, xn_h);

  // 3. in-projection GEMM (WMMA, N=1152, 16x64 strips)
  gemm_in_kernel<<<(NROWS / 16) * (NPROJ / 64) / 8, 256, 0, stream>>>(xn_h, w_in_h, zx);

  // 4. dt columns + softplus + log-decay
  dt_kernel<<<(NROWS * NH + 255) / 256, 256, 0, stream>>>(xn_h, w_in_h, dt_bias, A_log, dtv, lav);

  // 5. depthwise conv + silu
  conv_kernel<<<(int)(((size_t)NROWS * CONVD + 255) / 256), 256, 0, stream>>>(
      zx, conv_w, conv_b, xh_h, bm_h, cm_h);

  // 6. chunked SSM scan (WMMA), one block per (batch, head)
  scan_kernel<<<8 * NH, 256, 0, stream>>>(xh_h, bm_h, cm_h, dtv, lav, D_param, ys);

  // 7. gate + rmsnorm
  gate_rms_kernel<<<NROWS, 256, 0, stream>>>(ys, zx, rms_w, yg_h);

  // 8. out-projection GEMM + residual (WMMA)
  gemm_out_kernel<<<(NROWS / 16) * (DMODEL / 64) / 8, 256, 0, stream>>>(
      yg_h, w_out_h, x0, x1, sp_f, sp_h);

  // 9. MLP fc1 (concat via partner rows) + silu (WMMA)
  gemm_mlp1_kernel<<<(NROWS / 16) * (DMODEL / 64) / 8, 256, 0, stream>>>(sp_h, w1_h, mlp_b1, hm_h);

  // 10. MLP fc2 + bias + residual (WMMA)
  gemm_mlp2_kernel<<<(NROWS / 16) * (DMODEL / 64) / 8, 256, 0, stream>>>(hm_h, w2_h, mlp_b2, sp_f, u_f);

  // 11. final layernorm -> outputs (f0_fused then f1_fused, contiguous)
  final_ln_kernel<<<NROWS, 256, 0, stream>>>(u_f, nt_w, nt_b, out);
}
